// ILCBaseNode_25701084299369
// MI455X (gfx1250) — matrix operand order repaired and checked by reading, not compiled, not run on hardware
//
#include <hip/hip_runtime.h>

#define T_STEPS 32
#define BATCH   512
#define ACT     64
#define DEC     64
#define OUT_POP (ACT * DEC)

typedef __attribute__((ext_vector_type(16))) _Float16 v16h;
typedef __attribute__((ext_vector_type(8)))  _Float16 v8h;
typedef __attribute__((ext_vector_type(8)))  float    v8f;
typedef __attribute__((ext_vector_type(4)))  float    v4f;

__device__ __forceinline__ v16h cat8(v8h lo, v8h hi) {
  v16h r;
#pragma unroll
  for (int i = 0; i < 8; ++i) { r[i] = lo[i]; r[i + 8] = hi[i]; }
  return r;
}

// One block = (group a, tile of 16 batch rows). 4 waves; wave w owns output
// columns [16w, 16w+16). Recurrence over T runs inside the block.
__global__ __launch_bounds__(128, 2) void snn_if_decoder_kernel(
    const float* __restrict__ x_seq,    // [T, B, OUT_POP]
    const float* __restrict__ conn_w,   // [ACT, DEC, DEC]  (a, o, d)
    const float* __restrict__ conn_b,   // [OUT_POP]
    float* __restrict__ out)            // [T, B, OUT_POP] spikes
{
  __shared__ _Float16 sWhi[DEC * DEC];  // [o][d] f16 high part of W
  __shared__ _Float16 sWlo[DEC * DEC];  // [o][d] f16 residual of W
  __shared__ _Float16 sS[16 * DEC];     // spikes f16 [row][d]
  __shared__ float    sFB[16 * DEC];    // feedback [row][o]

  const int tid  = threadIdx.x;
  const int lane = tid & 31;
  const int wave = tid >> 5;                  // 0..3 -> output column chunk
  const int grp  = blockIdx.x & (ACT - 1);    // group a
  const int b0   = (blockIdx.x >> 6) * 16;    // batch tile base

  // ---- stage W for this group into LDS, split into f16 hi + residual ----
  const float* wsrc = conn_w + (size_t)grp * (DEC * DEC);
  for (int i = tid; i < DEC * DEC; i += 128) {
    float w      = wsrc[i];
    _Float16 hi  = (_Float16)w;
    sWhi[i]      = hi;
    sWlo[i]      = (_Float16)(w - (float)hi);
  }
  for (int i = tid; i < 16 * DEC; i += 128) sFB[i] = 0.0f;  // fb0 = 0
  __syncthreads();

  // ---- per-wave constant B fragments (held in VGPRs across all T steps) ----
  // B k order used here (contiguous, same as A): lane n = column; halves h=0..15 are
  // K = kbase + 16*(lane>>4) + h  -> contiguous in sW*[o][d].
  const int n    = lane & 15;
  const int hsel = lane >> 4;
  const int ocol = wave * 16 + n;
  const float bias = conn_b[grp * DEC + ocol];

  const _Float16* whrow = &sWhi[ocol * DEC + 16 * hsel];
  const _Float16* wlrow = &sWlo[ocol * DEC + 16 * hsel];
  v16h bhi0 = cat8(*(const v8h*)(whrow),      *(const v8h*)(whrow + 8));   // K 0..31
  v16h bhi1 = cat8(*(const v8h*)(whrow + 32), *(const v8h*)(whrow + 40));  // K 32..63
  v16h blo0 = cat8(*(const v8h*)(wlrow),      *(const v8h*)(wlrow + 8));
  v16h blo1 = cat8(*(const v8h*)(wlrow + 32), *(const v8h*)(wlrow + 40));

  // ---- per-thread neuron state: 8 channels, fixed (row, d) mapping ----
  const int row   = tid >> 3;
  const int dbase = (tid & 7) * 8;
  float v[8];
#pragma unroll
  for (int i = 0; i < 8; ++i) v[i] = 0.0f;

  const size_t gstep = (size_t)BATCH * OUT_POP;
  const size_t gbase = (size_t)(b0 + row) * OUT_POP + (size_t)grp * DEC + dbase;
  const int m = lane & 15;  // A-fragment row

  for (int t = 0; t < T_STEPS; ++t) {
    const float* xp = x_seq + (size_t)t * gstep + gbase;
    float*       op = out   + (size_t)t * gstep + gbase;
    if (t + 1 < T_STEPS) __builtin_prefetch(xp + gstep, 0, 0);  // global_prefetch_b8

    // -------- IF charge + fire + hard reset (v stays in registers) --------
    v4f x0 = *(const v4f*)xp;
    v4f x1 = *(const v4f*)(xp + 4);
    float sp[8];
#pragma unroll
    for (int i = 0; i < 8; ++i) {
      float xi = (i < 4) ? x0[i] : x1[i - 4];
      float vv = v[i] + xi + sFB[row * DEC + dbase + i];
      float s  = (vv >= 1.0f) ? 1.0f : 0.0f;   // rect(v - V_TH)
      v[i]     = vv - s * vv;                   // hard reset to 0
      sp[i]    = s;
      sS[row * DEC + dbase + i] = (_Float16)s;  // exact in f16
    }
    v4f s0 = {sp[0], sp[1], sp[2], sp[3]};
    v4f s1 = {sp[4], sp[5], sp[6], sp[7]};
    *(v4f*)op       = s0;
    *(v4f*)(op + 4) = s1;
    __syncthreads();

    // -------- grouped matmul: FB = S(16x64) * W^T(64x64) + bias ----------
    // A k order used here (contiguous, same as B): lanes 0-15 -> K {kb..kb+15},
    // lanes 16-31 -> K {kb+16..kb+31}, row m = lane&15.
    const _Float16* srow = &sS[m * DEC];
    v16h a0 = cat8(*(const v8h*)(srow + 16 * hsel),
                   *(const v8h*)(srow + 8 + 16 * hsel));        // K 0..31
    v16h a1 = cat8(*(const v8h*)(srow + 32 + 16 * hsel),
                   *(const v8h*)(srow + 40 + 16 * hsel));       // K 32..63

    v8f acc;
#pragma unroll
    for (int r = 0; r < 8; ++r) acc[r] = bias;   // bias depends only on column
    acc = __builtin_amdgcn_wmma_f32_16x16x32_f16(false, a0, false, bhi0,
                                                 (short)0, acc, false, false);
    acc = __builtin_amdgcn_wmma_f32_16x16x32_f16(false, a1, false, bhi1,
                                                 (short)0, acc, false, false);
    acc = __builtin_amdgcn_wmma_f32_16x16x32_f16(false, a0, false, blo0,
                                                 (short)0, acc, false, false);
    acc = __builtin_amdgcn_wmma_f32_16x16x32_f16(false, a1, false, blo1,
                                                 (short)0, acc, false, false);

    // D tile: VGPR r -> row r + 8*hsel, column = ocol
#pragma unroll
    for (int r = 0; r < 8; ++r) sFB[(r + 8 * hsel) * DEC + ocol] = acc[r];
    __syncthreads();
  }
}

extern "C" void kernel_launch(void* const* d_in, const int* in_sizes, int n_in,
                              void* d_out, int out_size, void* d_ws, size_t ws_size,
                              hipStream_t stream) {
  (void)in_sizes; (void)n_in; (void)out_size; (void)d_ws; (void)ws_size;
  const float* x_seq  = (const float*)d_in[0];
  const float* conn_w = (const float*)d_in[1];
  const float* conn_b = (const float*)d_in[2];
  float* out = (float*)d_out;

  // 64 groups * 32 batch tiles = 2048 blocks of 128 threads (4 waves)
  snn_if_decoder_kernel<<<dim3(ACT * (BATCH / 16)), dim3(128), 0, stream>>>(
      x_seq, conn_w, conn_b, out);
}
